// GHN_26371099197967
// MI455X (gfx1250) — matrix operand order repaired and checked by reading, not compiled, not run on hardware
//
#include <hip/hip_runtime.h>

// ---------------- problem constants ----------------
#define BSZ    8
#define NNODE  2048
#define HID    768
#define NH     4
#define BN     (BSZ * NNODE)          // 16384 rows
#define ALPHA  0.2f
#define NEGV   (-9.0e15f)

// ---------------- types ----------------
typedef unsigned short u16;
typedef __attribute__((ext_vector_type(8)))  unsigned short u16x8;
typedef __attribute__((ext_vector_type(16))) __bf16         v16bf;
typedef __attribute__((ext_vector_type(8)))  float          v8f;

struct Frag16 { u16x8 lo, hi; };   // 32B == one v16bf fragment

__device__ __forceinline__ u16 f2bf(float f) {
  unsigned u = __builtin_bit_cast(unsigned, f);
  u += 0x7FFFu + ((u >> 16) & 1u);               // round-to-nearest-even
  return (u16)(u >> 16);
}
__device__ __forceinline__ float bf2f(u16 h) {
  unsigned u = ((unsigned)h) << 16;
  return __builtin_bit_cast(float, u);
}
__device__ __forceinline__ v16bf frag_cast(Frag16 f) {
  return __builtin_bit_cast(v16bf, f);
}

// gfx1250 async Global->LDS copy (ASYNCcnt-tracked). LDS byte offset is the
// low 32 bits of the generic pointer (LDS aperture keeps the offset in [31:0]).
__device__ __forceinline__ void async_b128(const u16* g, u16* l) {
  unsigned loff = (unsigned)(unsigned long long)l;
  asm volatile("global_load_async_to_lds_b128 %0, %1, off"
               :: "v"(loff), "v"(g)
               : "memory");
}
#define WAIT_ASYNC(n) asm volatile("s_wait_asynccnt " #n ::: "memory")
#define WAIT_DS0      asm volatile("s_wait_dscnt 0x0" ::: "memory")

// Raw split workgroup barrier: no implicit loadcnt drain (unlike __syncthreads),
// so in-flight global loads legally survive across it.
__device__ __forceinline__ void wg_barrier() {
  asm volatile("s_barrier_signal -1\n\ts_barrier_wait -1" ::: "memory");
}

// ---------------- 1) weight convert + transpose: W (Z,K,N) f32 -> Wt (Z,N,K) bf16 ----------------
__global__ void k_cvt_w(const float* __restrict__ W, u16* __restrict__ Wt,
                        int K, int Nout, int total) {
  int tid = blockIdx.x * 256 + threadIdx.x;
  if (tid >= total) return;
  int z   = tid / (K * Nout);
  int rem = tid - z * (K * Nout);
  int k   = rem / Nout;
  int n   = rem - k * Nout;
  Wt[(long long)z * K * Nout + (long long)n * K + k] = f2bf(W[tid]);
}

// ---------------- 2) embedding gather -> bf16 A matrix ----------------
__global__ void k_gather(const int* __restrict__ nf, const float* __restrict__ emb,
                         u16* __restrict__ x, int total) {
  int tid = blockIdx.x * 256 + threadIdx.x;
  if (tid >= total) return;
  int i = tid / HID, f = tid - i * HID;
  x[tid] = f2bf(emb[nf[i] * HID + f]);
}

// ---------------- 3) GEMM: Wh = A(M,K) * B(K,768), output stored TRANSPOSED bf16 ----------------
// Async-depth-2 pipeline: DMA -> 4 LDS buffers -> register double-buffered B
// fragments -> WMMA. grid: (M/128, 768/64, Z); block 256 = 8 waves, 16x64/wave.
__device__ __forceinline__ void g_loadA(Frag16& fa, const u16* Abase, int k, int koffA) {
  fa.lo = *(const u16x8*)(Abase + k + koffA);
  fa.hi = *(const u16x8*)(Abase + k + 8 + koffA);
}
__device__ __forceinline__ void g_preload(Frag16* fb, const u16* buf, int lane, int koffB) {
#pragma unroll
  for (int t = 0; t < 4; ++t) {
    const u16* pb = &buf[(t * 16 + (lane & 15)) * 32 + koffB];
    fb[t].lo = *(const u16x8*)pb;
    fb[t].hi = *(const u16x8*)(pb + 8);
  }
}
__device__ __forceinline__ void g_wmma4(Frag16 fa, const Frag16* fb, v8f* c) {
  v16bf av = frag_cast(fa);
#pragma unroll
  for (int t = 0; t < 4; ++t)
    c[t] = __builtin_amdgcn_wmma_f32_16x16x32_bf16(
        false, av, false, frag_cast(fb[t]), (short)0, c[t], false, false);
}

__global__ void __launch_bounds__(256)
k_gemm(const u16* __restrict__ A, int K,
       const u16* __restrict__ Bt, long long bt_zstride,
       u16* __restrict__ OutT, long long out_zstride) {
  __shared__ __align__(16) u16 ldsB[4][64 * 32];   // 4-deep DMA pipeline (16KB)

  int lane = threadIdx.x & 31;
  int wave = threadIdx.x >> 5;
  int z  = blockIdx.z;
  int m0 = blockIdx.x * 128 + wave * 16;
  int f0 = blockIdx.y * 64;
  const u16* Bz = Bt + (long long)z * bt_zstride;

  // async staging role: thread t -> row t/4 (feature), 16B chunk t%4
  int srow   = threadIdx.x >> 2;
  int schunk = (threadIdx.x & 3) * 8;
  int soff   = srow * 32 + schunk;
  const u16* sbase = Bz + (long long)(f0 + srow) * K + schunk;
  u16* lds0 = &ldsB[0][0];

  v8f c[4] = {};
  int arow  = m0 + (lane & 15);
  int koffA = (lane >> 4) * 16;    // contiguous k order (matches B)
  int koffB = (lane >> 4) * 16;    // contiguous k order (matches A), column per lane
  const u16* Abase = A + (long long)arow * K;

  int nk = K / 32;                 // 24 or 96 (even, >=8)
  // prologue: stage tiles 0 and 1, preload tile 0 into registers
  async_b128(sbase,      lds0 + soff);             // tile0 -> L0
  async_b128(sbase + 32, lds0 + 2048 + soff);      // tile1 -> L1
  WAIT_ASYNC(1);                                   // tile0 landed
  wg_barrier();
  Frag16 R0[4], R1[4];
  g_preload(R0, lds0, lane, koffB);
  Frag16 fa0, fa1;
  g_loadA(fa0, Abase, 0, koffA);

  int t = 0;
  for (; t + 3 < nk; t += 2) {
    // ---- process tile t (R0/fa0); stage t+2; preload t+1 -> R1/fa1 ----
    async_b128(sbase + (t + 2) * 32, lds0 + ((t + 2) & 3) * 2048 + soff);
    WAIT_ASYNC(1);                                 // tile t+1 landed; t+2 in flight
    wg_barrier();
    g_preload(R1, lds0 + ((t + 1) & 3) * 2048, lane, koffB);
    g_loadA(fa1, Abase, (t + 1) * 32, koffA);
    g_wmma4(fa0, R0, c);                           // overlaps R1/fa1 loads

    // ---- process tile t+1 (R1/fa1); stage t+3; preload t+2 -> R0/fa0 ----
    async_b128(sbase + (t + 3) * 32, lds0 + ((t + 3) & 3) * 2048 + soff);
    WAIT_ASYNC(1);
    wg_barrier();
    g_preload(R0, lds0 + ((t + 2) & 3) * 2048, lane, koffB);
    g_loadA(fa0, Abase, (t + 2) * 32, koffA);
    g_wmma4(fa1, R1, c);
  }
  // epilogue: t == nk-2; tile nk-1 DMA outstanding
  WAIT_ASYNC(0);
  wg_barrier();
  g_preload(R1, lds0 + ((t + 1) & 3) * 2048, lane, koffB);
  g_loadA(fa1, Abase, (t + 1) * 32, koffA);
  g_wmma4(fa0, R0, c);                             // tile nk-2
  g_wmma4(fa1, R1, c);                             // tile nk-1

  // transposed store: lane holds 8 consecutive nodes for a fixed feature
  int b     = m0 >> 11;
  int node0 = (m0 & (NNODE - 1)) + (lane >> 4) * 8;
  u16* Oz = OutT + (long long)z * out_zstride + (long long)b * HID * NNODE;
#pragma unroll
  for (int tt = 0; tt < 4; ++tt) {
    int f = f0 + tt * 16 + (lane & 15);
    u16x8 u;
#pragma unroll
    for (int r = 0; r < 8; ++r) u[r] = f2bf(c[tt][r]);
    *(u16x8*)(Oz + (long long)f * NNODE + node0) = u;
  }
}

// ---------------- 4) Wh1/Wh2 = Wh . a halves (from feature-major WhT) ----------------
// grid: (2048/256, B, Z)
__global__ void k_avec(const u16* __restrict__ WhT, long long whz,
                       const float* __restrict__ a, long long az,
                       float* __restrict__ w1, float* __restrict__ w2,
                       long long wz, int F) {
  int z = blockIdx.z, b = blockIdx.y;
  int node = blockIdx.x * 256 + threadIdx.x;
  const u16*   base = WhT + (long long)z * whz + (long long)b * F * NNODE + node;
  const float* av   = a + (long long)z * az;
  float s1 = 0.f, s2 = 0.f;
  for (int f = 0; f < F; ++f) {
    float v = bf2f(base[(long long)f * NNODE]);
    s1 += v * av[f];
    s2 += v * av[F + f];
  }
  long long o = (long long)z * wz + (long long)b * NNODE + node;
  w1[o] = s1;
  w2[o] = s2;
}

// ---------------- 5) fused GAT attention: online softmax + flash-style att@Wh ----------------
// grid: (N/16, B, Z); block 256 = 8 waves; wave owns a 96-feature slice.
// 2x-unrolled pipeline, named double buffers; raw barriers keep the 12
// B-fragment global loads in flight across tile boundaries.
__global__ void __launch_bounds__(256)
k_gat_attn(const int* __restrict__ adj,
           const float* __restrict__ w1, const float* __restrict__ w2, long long wz,
           const u16* __restrict__ WhT, long long whz,
           u16* __restrict__ out_bf,    // layer 1: concat dest (stride 3072), else null
           float* __restrict__ out_f32, // layer 2: f32 dest (stride 768), else null
           int out_stride) {
  __shared__ __align__(16) u16 pbuf[2][16 * 32];
  __shared__ float sm_m[16], sm_li[16];

  int lane = threadIdx.x & 31;
  int wave = threadIdx.x >> 5;
  int z = blockIdx.z, b = blockIdx.y;
  int m0 = blockIdx.x * 16;

  const float* W1 = w1 + (long long)z * wz + (long long)b * NNODE;
  const float* W2 = w2 + (long long)z * wz + (long long)b * NNODE;
  const int* adjb = adj + ((long long)b * NNODE + m0) * NNODE;
  const u16* whb  = WhT + (long long)z * whz + (long long)b * HID * NNODE;

  // ---- phase 1: per-row online softmax stats (masked rows keep NEGV ->
  //      all-masked rows degrade to the reference's uniform softmax) ----
  for (int rr = 0; rr < 2; ++rr) {
    int row = wave * 2 + rr;
    const int* adjr = adjb + (long long)row * NNODE;
    float wh1 = W1[m0 + row];
    float m = -3.0e38f, l = 0.f;
    for (int j = lane; j < NNODE; j += 32) {
      __builtin_prefetch(adjr + j + 512, 0, 1);
      float v = wh1 + W2[j];
      float e = v > 0.f ? v : ALPHA * v;
      if (adjr[j] <= 0) e = NEGV;
      float mn = fmaxf(m, e);
      l = l * __expf(m - mn) + __expf(e - mn);
      m = mn;
    }
#pragma unroll
    for (int off = 16; off > 0; off >>= 1) {
      float mo = __shfl_xor(m, off, 32);
      float lo = __shfl_xor(l, off, 32);
      float mn = fmaxf(m, mo);
      l = l * __expf(m - mn) + lo * __expf(mo - mn);
      m = mn;
    }
    if (lane == 0) { sm_m[row] = m; sm_li[row] = 1.0f / l; }
  }
  WAIT_DS0;          // publish sm_m / sm_li
  wg_barrier();

  // per-thread p-tile roles (two elements: rows r0 and r0+8, column pcol)
  int r0 = threadIdx.x >> 5;         // 0..7
  int pcol = threadIdx.x & 31;
  float wh1_0 = W1[m0 + r0], wh1_1 = W1[m0 + r0 + 8];
  float mA = sm_m[r0],     liA = sm_li[r0];
  float mB = sm_m[r0 + 8], liB = sm_li[r0 + 8];
  const int* adjr0 = adjb + (long long)r0 * NNODE;
  const int* adjr1 = adjb + (long long)(r0 + 8) * NNODE;

  int f0    = wave * 96;
  int arowA = lane & 15;
  int koffA = (lane >> 4) * 16;
  int koffB = (lane >> 4) * 16;

  auto compute_p = [&](int jt, int buf) {
    int j = jt * 32 + pcol;
    float w2j = W2[j];
    float v = wh1_0 + w2j;
    float e = v > 0.f ? v : ALPHA * v;
    if (adjr0[j] <= 0) e = NEGV;
    pbuf[buf][r0 * 32 + pcol] = f2bf(__expf(e - mA) * liA);
    v = wh1_1 + w2j;
    e = v > 0.f ? v : ALPHA * v;
    if (adjr1[j] <= 0) e = NEGV;
    pbuf[buf][(r0 + 8) * 32 + pcol] = f2bf(__expf(e - mB) * liB);
  };
  auto load_b = [&](int jt, Frag16* fb) {
    int j0 = jt * 32 + koffB;
#pragma unroll
    for (int t = 0; t < 6; ++t) {
      const u16* pb = whb + (long long)(f0 + t * 16 + (lane & 15)) * NNODE + j0;
      fb[t].lo = *(const u16x8*)pb;
      fb[t].hi = *(const u16x8*)(pb + 8);
    }
  };
  auto read_a = [&](int buf) -> Frag16 {
    Frag16 fa;
    fa.lo = *(const u16x8*)&pbuf[buf][arowA * 32 + koffA];
    fa.hi = *(const u16x8*)&pbuf[buf][arowA * 32 + 8 + koffA];
    return fa;
  };

  v8f c[6] = {};
  Frag16 fb0[6], fb1[6];
  compute_p(0, 0);            // prologue
  load_b(0, fb0);

  int jt = 0;
  for (; jt + 2 < NNODE / 32; jt += 2) {
    WAIT_DS0;                              // publish p-tile writes
    wg_barrier();                          // pbuf0 written; pbuf1 readers done
    Frag16 fa = read_a(0);
    load_b(jt + 1, fb1);                   // stays in flight under WMMAs
    compute_p(jt + 1, 1);
    v16bf av = frag_cast(fa);
#pragma unroll
    for (int t = 0; t < 6; ++t)
      c[t] = __builtin_amdgcn_wmma_f32_16x16x32_bf16(
          false, av, false, frag_cast(fb0[t]), (short)0, c[t], false, false);

    WAIT_DS0;
    wg_barrier();                          // pbuf1 written; pbuf0 readers done
    fa = read_a(1);
    load_b(jt + 2, fb0);
    compute_p(jt + 2, 0);
    av = frag_cast(fa);
#pragma unroll
    for (int t = 0; t < 6; ++t)
      c[t] = __builtin_amdgcn_wmma_f32_16x16x32_bf16(
          false, av, false, frag_cast(fb1[t]), (short)0, c[t], false, false);
  }
  // epilogue: tiles 62 (pbuf0/fb0) and 63 (pbuf1/fb1)
  {
    WAIT_DS0;
    wg_barrier();
    Frag16 fa = read_a(0);
    load_b(jt + 1, fb1);
    compute_p(jt + 1, 1);
    v16bf av = frag_cast(fa);
#pragma unroll
    for (int t = 0; t < 6; ++t)
      c[t] = __builtin_amdgcn_wmma_f32_16x16x32_bf16(
          false, av, false, frag_cast(fb0[t]), (short)0, c[t], false, false);
    WAIT_DS0;
    wg_barrier();
    fa = read_a(1);
    av = frag_cast(fa);
#pragma unroll
    for (int t = 0; t < 6; ++t)
      c[t] = __builtin_amdgcn_wmma_f32_16x16x32_bf16(
          false, av, false, frag_cast(fb1[t]), (short)0, c[t], false, false);
  }

  // ---- epilogue: ELU, store (bf16 concat for layer 1, f32 for layer 2) ----
#pragma unroll
  for (int t = 0; t < 6; ++t) {
    int f = f0 + t * 16 + (lane & 15);
#pragma unroll
    for (int r = 0; r < 8; ++r) {
      int node = m0 + (lane >> 4) * 8 + r;
      float v = c[t][r];
      v = v > 0.f ? v : __expf(v) - 1.f;
      if (out_bf)
        out_bf[((long long)b * NNODE + node) * out_stride + z * HID + f] = f2bf(v);
      else
        out_f32[((long long)b * NNODE + node) * out_stride + f] = v;
    }
  }
}

// ---------------- 6) mean pool over nodes ----------------
__global__ void k_pool(const float* __restrict__ h2, float* __restrict__ out) {
  int tid = blockIdx.x * 256 + threadIdx.x;     // b*768 + f
  if (tid >= BSZ * HID) return;
  int b = tid / HID, f = tid - b * HID;
  const float* p = h2 + (long long)b * NNODE * HID + f;
  float s = 0.f;
  for (int n = 0; n < NNODE; ++n) s += p[(long long)n * HID];
  out[tid] = s * (1.0f / (float)NNODE);
}

// ---------------- launch ----------------
extern "C" void kernel_launch(void* const* d_in, const int* in_sizes, int n_in,
                              void* d_out, int out_size, void* d_ws, size_t ws_size,
                              hipStream_t stream) {
  const int*   node_feats = (const int*)d_in[0];
  const int*   adjs       = (const int*)d_in[1];
  const float* embed      = (const float*)d_in[2];
  const float* W_heads    = (const float*)d_in[3];
  const float* a_heads    = (const float*)d_in[4];
  const float* W_out      = (const float*)d_in[5];
  const float* a_out      = (const float*)d_in[6];
  float* out = (float*)d_out;

  char* ws = (char*)d_ws;
  size_t off = 0;
  auto take = [&](size_t bytes) -> char* {
    char* p = ws + off;
    off = (off + bytes + 255) & ~(size_t)255;
    return p;
  };

  u16*   x_bf = (u16*)take((size_t)BN * HID * 2);                 // embeddings, bf16
  u16*   wtH  = (u16*)take((size_t)NH * HID * HID * 2);           // W_heads^T bf16
  u16*   wtO  = (u16*)take((size_t)(NH * HID) * HID * 2);         // W_out^T bf16
  u16*   whT1 = (u16*)take((size_t)NH * BSZ * HID * NNODE * 2);   // layer-1 Wh^T (per head)
  float* w1a  = (float*)take((size_t)NH * BN * 4);
  float* w2a  = (float*)take((size_t)NH * BN * 4);
  u16*   x1   = (u16*)take((size_t)BN * (NH * HID) * 2);          // concat heads bf16
  u16*   whT2 = (u16*)take((size_t)BSZ * HID * NNODE * 2);        // layer-2 Wh^T
  float* w1o  = (float*)take((size_t)BN * 4);
  float* w2o  = (float*)take((size_t)BN * 4);
  float* h2   = (float*)take((size_t)BN * HID * 4);               // layer-2 output f32

  // weights -> transposed bf16
  {
    int total = NH * HID * HID;
    k_cvt_w<<<(total + 255) / 256, 256, 0, stream>>>(W_heads, wtH, HID, HID, total);
  }
  {
    int total = (NH * HID) * HID;
    k_cvt_w<<<(total + 255) / 256, 256, 0, stream>>>(W_out, wtO, NH * HID, HID, total);
  }
  // embedding gather
  {
    int total = BN * HID;
    k_gather<<<(total + 255) / 256, 256, 0, stream>>>(node_feats, embed, x_bf, total);
  }
  // layer 1: Wh per head (transposed), a-dots, fused attention -> concat x1
  k_gemm<<<dim3(BN / 128, HID / 64, NH), 256, 0, stream>>>(
      x_bf, HID, wtH, (long long)HID * HID, whT1, (long long)BSZ * HID * NNODE);
  k_avec<<<dim3(NNODE / 256, BSZ, NH), 256, 0, stream>>>(
      whT1, (long long)BSZ * HID * NNODE, a_heads, 2LL * HID,
      w1a, w2a, (long long)BN, HID);
  k_gat_attn<<<dim3(NNODE / 16, BSZ, NH), 256, 0, stream>>>(
      adjs, w1a, w2a, (long long)BN, whT1, (long long)BSZ * HID * NNODE,
      x1, nullptr, NH * HID);
  // layer 2: output GEMM (K=3072), a-dots, fused attention -> h2 (f32, ELU applied)
  k_gemm<<<dim3(BN / 128, HID / 64, 1), 256, 0, stream>>>(
      x1, NH * HID, wtO, 0LL, whT2, 0LL);
  k_avec<<<dim3(NNODE / 256, BSZ, 1), 256, 0, stream>>>(
      whT2, 0LL, a_out, 0LL, w1o, w2o, 0LL, HID);
  k_gat_attn<<<dim3(NNODE / 16, BSZ, 1), 256, 0, stream>>>(
      adjs, w1o, w2o, 0LL, whT2, 0LL, nullptr, h2, HID);
  // mean pool
  k_pool<<<(BSZ * HID + 255) / 256, 256, 0, stream>>>(h2, out);
}
